// SlidingWindowMHA_8615704396092
// MI455X (gfx1250) — matrix operand order repaired and checked by reading, not compiled, not run on hardware
//
#include <hip/hip_runtime.h>
#include <hip/hip_bf16.h>

// Sliding-window MHA for MI455X (gfx1250, wave32, WMMA).
// B=2, T=2048, D=512, H=8, C=64, WIN=128.  All GEMMs on v_wmma_f32_16x16x32_bf16.

#define BB  2
#define TT  2048
#define DD  512
#define HHH 8
#define CCC 64
#define HCC 512
#define WWIN 128

typedef __attribute__((ext_vector_type(16))) __bf16 v16bf;
typedef __attribute__((ext_vector_type(4)))  __bf16 v4bf;
typedef __attribute__((ext_vector_type(8)))  float  v8f;

#define WMMA_BF16(a, b, c) \
  __builtin_amdgcn_wmma_f32_16x16x32_bf16(false, (a), false, (b), (short)0, (c), false, false)

#define V8F_ZERO {0.f,0.f,0.f,0.f,0.f,0.f,0.f,0.f}

static __device__ __forceinline__ v4bf cvt4(const float4 f) {
  v4bf r;
  r[0] = (__bf16)f.x; r[1] = (__bf16)f.y; r[2] = (__bf16)f.z; r[3] = (__bf16)f.w;
  return r;
}

// ---------------------------------------------------------------------------
// Kernel 1: QKV projection (bf16 WMMA) + per-head RMS norm + RoPE.
// Workgroup = 32 rows of h. 8 waves: wave = (mt in 0..1) x (nc in 0..3).
// Each wave owns a 16x64 output tile; 64 cols == exactly one head.
// ---------------------------------------------------------------------------
__global__ __launch_bounds__(256) void qkv_rope_kernel(
    const float* __restrict__ h,
    const float* __restrict__ rope_cos,
    const float* __restrict__ rope_sin,
    const float* __restrict__ Wq,
    const float* __restrict__ Wk,
    const float* __restrict__ Wv,
    const float* __restrict__ qnw,
    const float* __restrict__ knw,
    __bf16* __restrict__ qo,
    __bf16* __restrict__ ko,
    __bf16* __restrict__ vo)
{
  __shared__ __bf16 sh_h[32][DD];     // 32 KB: h tile (bf16)
  __shared__ __bf16 sh_w[256][32];    // 16 KB: W chunk transposed; reused as store bounce

  const int tid  = threadIdx.x;
  const int row0 = blockIdx.x * 32;   // global row in [0, B*T)

  // Vectorized h staging: coalesced float4 loads -> v4bf (b64) LDS stores.
  {
    const float4* __restrict__ hsrc = (const float4*)(h + (size_t)row0 * DD);
    v4bf* __restrict__ hdst = (v4bf*)&sh_h[0][0];
#pragma unroll
    for (int it = 0; it < 16; ++it) {
      const int i = tid + it * 256;           // 0..4095 float4 groups
      hdst[i] = cvt4(hsrc[i]);
    }
  }
  __syncthreads();

  const int wave = tid >> 5, lane = tid & 31;
  const int mt = wave >> 2, nc = wave & 3;
  const int l16 = lane & 15, lh = lane >> 4;
  const v8f vzero = V8F_ZERO;

#pragma unroll
  for (int mat = 0; mat < 3; ++mat) {
    const float* __restrict__ W    = (mat == 0) ? Wq : (mat == 1) ? Wk : Wv;
    __bf16*      __restrict__ outp = (mat == 0) ? qo : (mat == 1) ? ko : vo;

#pragma unroll
    for (int np = 0; np < 2; ++np) {
      const int n0 = np * 256 + nc * 64;      // wave's column base (head boundary)
      v8f acc[4] = {vzero, vzero, vzero, vzero};

      for (int kk = 0; kk < DD; kk += 32) {
        __syncthreads();
        // Stage W[kk:kk+32, np*256:+256] transposed into LDS via 4x4 register
        // tiles: float4 (b128) coalesced loads, v4bf (b64) LDS stores.
#pragma unroll
        for (int it = 0; it < 2; ++it) {
          const int tile = tid + it * 256;    // 0..511
          const int c0 = (tile & 63) * 4;     // 0..252
          const int r0 = (tile >> 6) * 4;     // 0..28
          const int kp = (kk + 32 < DD) ? (kk + 32) : kk;   // branchless prefetch
          __builtin_prefetch(&W[(size_t)(kp + r0) * HCC + np * 256 + c0], 0, 1);
          v4bf rows[4];
#pragma unroll
          for (int t = 0; t < 4; ++t)
            rows[t] = cvt4(*(const float4*)&W[(size_t)(kk + r0 + t) * HCC + np * 256 + c0]);
#pragma unroll
          for (int c = 0; c < 4; ++c) {
            v4bf q4 = {rows[0][c], rows[1][c], rows[2][c], rows[3][c]};
            *(v4bf*)&sh_w[c0 + c][r0] = q4;
          }
        }
        __syncthreads();

        // A (16x32 bf16): lane row = mt*16 + lane%16; K halves per lane group.
        v16bf a;
        {
          const int ar = mt * 16 + l16;
          const int kb = kk + lh * 16;
#pragma unroll
          for (int e = 0; e < 16; ++e) a[e] = sh_h[ar][kb + e];
        }
#pragma unroll
        for (int j = 0; j < 4; ++j) {
          // B (32x16 bf16): 16 contiguous K elems per lane -> 2x ds_load_b128.
          v16bf bm;
          const int bc  = nc * 64 + j * 16 + l16;
          const int kb2 = lh * 16;
#pragma unroll
          for (int e = 0; e < 16; ++e) bm[e] = sh_w[bc][kb2 + e];
          acc[j] = WMMA_BF16(a, bm, acc[j]);
        }
      }

      if (mat < 2) {
        // ---- RMS norm over the head's 64 channels (per row) ----
        const float* nw = (mat == 0) ? qnw : knw;
        float wc[4];
#pragma unroll
        for (int j = 0; j < 4; ++j) wc[j] = nw[j * 16 + l16];
#pragma unroll
        for (int r = 0; r < 8; ++r) {
          float ss = 0.f;
#pragma unroll
          for (int j = 0; j < 4; ++j) ss += acc[j][r] * acc[j][r];
#pragma unroll
          for (int m = 1; m < 16; m <<= 1) ss += __shfl_xor(ss, m, 32);
          const float nrm = rsqrtf(ss * (1.0f / 64.0f) + 1e-6f);
#pragma unroll
          for (int j = 0; j < 4; ++j) acc[j][r] *= nrm * wc[j];
        }
        // ---- RoPE (rotation partner is +/-32 cols == acc[(j+2)%4], same lane) ----
#pragma unroll
        for (int r = 0; r < 8; ++r) {
          const int rg = row0 + mt * 16 + r + lh * 8;
          const int t  = rg & (TT - 1);
          float x[4], cs[4], sn[4];
#pragma unroll
          for (int j = 0; j < 4; ++j) {
            x[j] = acc[j][r];
            const int c = j * 16 + l16;
            cs[j] = rope_cos[t * CCC + c];
            sn[j] = rope_sin[t * CCC + c];
          }
#pragma unroll
          for (int j = 0; j < 4; ++j) {
            const float rot = (j < 2) ? -x[j + 2] : x[j - 2];
            acc[j][r] = x[j] * cs[j] + rot * sn[j];
          }
        }
      }

      // ---- store: bounce 16x64 tile through LDS -> coalesced b32 row stores ----
      __syncthreads();   // all waves done reading sh_w this K pass; reuse as bounce
      __bf16 (* __restrict__ st)[64] = (__bf16 (*)[64])(&sh_w[0][0] + wave * 1024);
#pragma unroll
      for (int r = 0; r < 8; ++r)
#pragma unroll
        for (int j = 0; j < 4; ++j)
          st[r + lh * 8][j * 16 + l16] = (__bf16)acc[j][r];
      __builtin_amdgcn_fence(__ATOMIC_ACQ_REL, "wavefront");
#pragma unroll
      for (int rr = 0; rr < 16; ++rr) {
        const unsigned int val = *(const unsigned int*)&st[rr][lane * 2];
        const int rg = row0 + mt * 16 + rr;
        *(unsigned int*)&outp[(size_t)rg * HCC + n0 + lane * 2] = val;
      }
    }
  }
}

// ---------------------------------------------------------------------------
// Kernel 2: sliding-window flash attention. One 16-query tile per wave, per
// (batch, head). QK^T and PV on bf16 WMMA; online softmax. V slab for the
// whole block is staged TRANSPOSED in LDS once, so PV B tiles are contiguous
// ds_load_b128. P transposed C->A layout via a per-wave LDS bounce.
// ---------------------------------------------------------------------------
__global__ __launch_bounds__(256) void attn_kernel(
    const __bf16* __restrict__ q,
    const __bf16* __restrict__ k,
    const __bf16* __restrict__ v,
    __bf16* __restrict__ ao)
{
  __shared__ __bf16 sh_vt[64][288];      // 36 KB: V^T slab (col x key)
  __shared__ __bf16 sp_all[8][16][32];   //  8 KB: per-wave P transpose scratch
  __shared__ __bf16 sh_ob[8][16][64];    // 16 KB: per-wave output bounce

  const int tid = threadIdx.x, wave = tid >> 5, lane = tid & 31;
  const int l16 = lane & 15, lh = lane >> 4;
  const int blk    = blockIdx.x;
  const int qchunk = blk & 15;
  const int hh     = (blk >> 4) & 7;
  const int b      = blk >> 7;
  const int qbase  = qchunk * 128 + wave * 16;
  const size_t base = ((size_t)b * TT) * HCC + hh * CCC;
  __bf16 (* __restrict__ sp)[32] = sp_all[wave];

  // Block-wide key slab: [kstart, kstart+288) covers every wave's window + PV tail.
  const int kstart = (qchunk == 0) ? 0 : (qchunk * 128 - 128);

  // Stage V^T via 4x4 register-transposed tiles (b64 loads, b64 stores).
  for (int tile = tid; tile < 1152; tile += 256) {   // 72x16 tiles of 4x4
    const int k0 = (tile % 72) * 4;
    const int c0 = (tile / 72) * 4;
    v4bf rows[4];
#pragma unroll
    for (int t = 0; t < 4; ++t) {
      int key = kstart + k0 + t;
      key = (key < TT) ? key : TT - 1;
      rows[t] = *(const v4bf*)&v[base + (size_t)key * HCC + c0];
    }
#pragma unroll
    for (int c = 0; c < 4; ++c) {
      v4bf q4 = {rows[0][c], rows[1][c], rows[2][c], rows[3][c]};
      *(v4bf*)&sh_vt[c0 + c][k0] = q4;
    }
  }

  // Q tile as two A operands (c 0..31 and 32..63) while V stages.
  v16bf aq0, aq1;
  {
    const size_t qrow = base + (size_t)(qbase + l16) * HCC;
    const int kb = lh * 16;
#pragma unroll
    for (int e = 0; e < 16; ++e) {
      aq0[e] = q[qrow + kb + e];
      aq1[e] = q[qrow + 32 + kb + e];
    }
  }
  __syncthreads();

  const v8f vzero = V8F_ZERO;
  v8f o[4] = {vzero, vzero, vzero, vzero};
  float mrow[8], lsum[8];
#pragma unroll
  for (int r = 0; r < 8; ++r) { mrow[r] = -1e30f; lsum[r] = 0.f; }

  int jstart = qbase - (WWIN - 1);
  jstart = (jstart < 0) ? 0 : (jstart & ~31);
  const int jend = qbase + 15;
  const int kb16 = lh * 16, kb8 = lh * 8;

  for (int j0 = jstart; j0 <= jend; j0 += 32) {
    v8f s0 = vzero, s1 = vzero;
    const int key0 = j0 + l16;
    const int key1 = key0 + 16;
    const size_t kr0 = base + (size_t)(key0 < TT ? key0 : TT - 1) * HCC;
    const size_t kr1 = base + (size_t)(key1 < TT ? key1 : TT - 1) * HCC;
    {
      v16bf bk;   // K^T B-operand: contiguous 32B per lane -> global_load_b128 x2
#pragma unroll
      for (int e = 0; e < 16; ++e) bk[e] = k[kr0 + kb16 + e];
      s0 = WMMA_BF16(aq0, bk, s0);
#pragma unroll
      for (int e = 0; e < 16; ++e) bk[e] = k[kr0 + 32 + kb16 + e];
      s0 = WMMA_BF16(aq1, bk, s0);
#pragma unroll
      for (int e = 0; e < 16; ++e) bk[e] = k[kr1 + kb16 + e];
      s1 = WMMA_BF16(aq0, bk, s1);
#pragma unroll
      for (int e = 0; e < 16; ++e) bk[e] = k[kr1 + 32 + kb16 + e];
      s1 = WMMA_BF16(aq1, bk, s1);
    }

    // Online softmax over this 32-key slab (row stats across 16-lane halves).
#pragma unroll
    for (int r = 0; r < 8; ++r) {
      const int trow = qbase + r + lh * 8;
      float x0 = (key0 <= trow && key0 > trow - WWIN) ? s0[r] * 0.125f : -1e30f;
      float x1 = (key1 <= trow && key1 > trow - WWIN) ? s1[r] * 0.125f : -1e30f;
      float mx = fmaxf(x0, x1);
#pragma unroll
      for (int m = 1; m < 16; m <<= 1) mx = fmaxf(mx, __shfl_xor(mx, m, 32));
      const float mnew = fmaxf(mrow[r], mx);
      const float corr = __expf(mrow[r] - mnew);
      const float p0 = __expf(x0 - mnew);
      const float p1 = __expf(x1 - mnew);
      float ps = p0 + p1;
#pragma unroll
      for (int m = 1; m < 16; m <<= 1) ps += __shfl_xor(ps, m, 32);
      lsum[r] = lsum[r] * corr + ps;
      mrow[r] = mnew;
#pragma unroll
      for (int j = 0; j < 4; ++j) o[j][r] *= corr;
      s0[r] = p0; s1[r] = p1;
    }

    // C-layout P -> A-layout via per-wave LDS (DScnt in-order within wave).
#pragma unroll
    for (int r = 0; r < 8; ++r) {
      sp[r + lh * 8][l16]      = (__bf16)s0[r];
      sp[r + lh * 8][16 + l16] = (__bf16)s1[r];
    }
    __builtin_amdgcn_fence(__ATOMIC_ACQ_REL, "wavefront");
    v16bf ap;
#pragma unroll
    for (int e = 0; e < 16; ++e) ap[e] = sp[l16][kb16 + e];

    // PV: B tiles from V^T in LDS -> contiguous 32B per lane (ds_load_b128 x2).
    const int jl = j0 - kstart + kb16;
#pragma unroll
    for (int jt = 0; jt < 4; ++jt) {
      v16bf bv;
      const int cn = jt * 16 + l16;
#pragma unroll
      for (int e = 0; e < 16; ++e) bv[e] = sh_vt[cn][jl + e];
      o[jt] = WMMA_BF16(ap, bv, o[jt]);
    }
  }

  // Normalize; bounce through LDS for coalesced b32 row stores.
  __bf16 (* __restrict__ st)[64] = sh_ob[wave];
#pragma unroll
  for (int r = 0; r < 8; ++r) {
    const float inv = 1.0f / lsum[r];
#pragma unroll
    for (int jt = 0; jt < 4; ++jt)
      st[r + lh * 8][jt * 16 + l16] = (__bf16)(o[jt][r] * inv);
  }
  __builtin_amdgcn_fence(__ATOMIC_ACQ_REL, "wavefront");
#pragma unroll
  for (int rr = 0; rr < 16; ++rr) {
    const unsigned int val = *(const unsigned int*)&st[rr][lane * 2];
    const int trow = qbase + rr;
    *(unsigned int*)&ao[base + (size_t)trow * HCC + lane * 2] = val;
  }
}

// ---------------------------------------------------------------------------
// Kernel 3: output projection attn(bf16) @ W_o -> fp32 out. Same GEMM skeleton.
// ---------------------------------------------------------------------------
__global__ __launch_bounds__(256) void oproj_kernel(
    const __bf16* __restrict__ a_in,
    const float* __restrict__ Wo,
    float* __restrict__ out)
{
  __shared__ __bf16 sh_a[32][DD];     // 32 KB
  __shared__ __bf16 sh_w[256][32];    // 16 KB

  const int tid  = threadIdx.x;
  const int row0 = blockIdx.x * 32;

  // bf16 input: straight 16B copies into LDS.
  {
    const uint4* __restrict__ asrc = (const uint4*)(a_in + (size_t)row0 * DD);
    uint4* __restrict__ adst = (uint4*)&sh_a[0][0];
#pragma unroll
    for (int it = 0; it < 8; ++it) {
      const int i = tid + it * 256;        // 0..2047 uint4 groups
      adst[i] = asrc[i];
    }
  }
  __syncthreads();

  const int wave = tid >> 5, lane = tid & 31;
  const int mt = wave >> 2, nc = wave & 3;
  const int l16 = lane & 15, lh = lane >> 4;
  const v8f vzero = V8F_ZERO;

#pragma unroll
  for (int np = 0; np < 2; ++np) {
    const int n0 = np * 256 + nc * 64;
    v8f acc[4] = {vzero, vzero, vzero, vzero};

    for (int kk = 0; kk < DD; kk += 32) {
      __syncthreads();
#pragma unroll
      for (int it = 0; it < 2; ++it) {
        const int tile = tid + it * 256;
        const int c0 = (tile & 63) * 4;
        const int r0 = (tile >> 6) * 4;
        const int kp = (kk + 32 < DD) ? (kk + 32) : kk;
        __builtin_prefetch(&Wo[(size_t)(kp + r0) * HCC + np * 256 + c0], 0, 1);
        v4bf rows[4];
#pragma unroll
        for (int t = 0; t < 4; ++t)
          rows[t] = cvt4(*(const float4*)&Wo[(size_t)(kk + r0 + t) * HCC + np * 256 + c0]);
#pragma unroll
        for (int c = 0; c < 4; ++c) {
          v4bf q4 = {rows[0][c], rows[1][c], rows[2][c], rows[3][c]};
          *(v4bf*)&sh_w[c0 + c][r0] = q4;
        }
      }
      __syncthreads();

      v16bf a;
      {
        const int ar = mt * 16 + l16;
        const int kb = kk + lh * 16;
#pragma unroll
        for (int e = 0; e < 16; ++e) a[e] = sh_a[ar][kb + e];
      }
#pragma unroll
      for (int j = 0; j < 4; ++j) {
        v16bf bm;
        const int bc  = nc * 64 + j * 16 + l16;
        const int kb2 = lh * 16;
#pragma unroll
        for (int e = 0; e < 16; ++e) bm[e] = sh_w[bc][kb2 + e];
        acc[j] = WMMA_BF16(a, bm, acc[j]);
      }
    }
#pragma unroll
    for (int r = 0; r < 8; ++r) {
      const int rg = row0 + mt * 16 + r + lh * 8;
#pragma unroll
      for (int j = 0; j < 4; ++j)
        out[(size_t)rg * HCC + n0 + j * 16 + l16] = acc[j][r];
    }
  }
}

// ---------------------------------------------------------------------------
extern "C" void kernel_launch(void* const* d_in, const int* in_sizes, int n_in,
                              void* d_out, int out_size, void* d_ws, size_t ws_size,
                              hipStream_t stream) {
  (void)in_sizes; (void)n_in; (void)out_size; (void)ws_size;

  const float* h        = (const float*)d_in[0];
  const float* rope_cos = (const float*)d_in[1];
  const float* rope_sin = (const float*)d_in[2];
  const float* Wq       = (const float*)d_in[3];
  const float* Wk       = (const float*)d_in[4];
  const float* Wv       = (const float*)d_in[5];
  const float* Wo       = (const float*)d_in[6];
  const float* qnw      = (const float*)d_in[7];
  const float* knw      = (const float*)d_in[8];
  float* out = (float*)d_out;

  const size_t N = (size_t)BB * TT * HCC;   // 2M elements per tensor
  __bf16* qb = (__bf16*)d_ws;               // 4 MB
  __bf16* kb = qb + N;                      // 4 MB
  __bf16* vb = kb + N;                      // 4 MB
  __bf16* ab = vb + N;                      // 4 MB  (16 MB workspace total)

  qkv_rope_kernel<<<dim3((BB * TT) / 32), dim3(256), 0, stream>>>(
      h, rope_cos, rope_sin, Wq, Wk, Wv, qnw, knw, qb, kb, vb);
  attn_kernel<<<dim3(BB * HHH * (TT / 128)), dim3(256), 0, stream>>>(
      qb, kb, vb, ab);
  oproj_kernel<<<dim3((BB * TT) / 32), dim3(256), 0, stream>>>(
      ab, Wo, out);
}
